// Rel_Attention_7215545057839
// MI455X (gfx1250) — matrix operand order repaired and checked by reading, not compiled, not run on hardware
//
#include <hip/hip_runtime.h>

// ---------------------------------------------------------------------------
// Rel-position window attention, CDNA5 (gfx1250), wave32 + bf16 WMMA.
// D = A x B + C via v_wmma_f32_16x16x32_bf16 with fp32 accumulation.
// ---------------------------------------------------------------------------

typedef __bf16 bf16;
typedef __bf16  v16bf __attribute__((ext_vector_type(16)));
typedef float   v8f   __attribute__((ext_vector_type(8)));

union FragBF { v16bf v; uint4 q[2]; };

// ---- problem constants ----
constexpr int   HEADS = 8;
constexpr int   DH    = 64;
constexpr int   DIM   = 512;
constexpr int   NSEQ  = 1025;          // w*w + 1
constexpr int   BATCH = 32;
constexpr int   NPAD  = 1056;          // 33 * 32, covers 1025 keys
constexpr int   QT2   = 33;            // ceil(1025/32) 32-query super-tiles
constexpr float EPS   = 1e-6f;
constexpr float SCALE = 0.125f;        // 64^-0.5

// ---------------------------------------------------------------------------
// Fragment loaders (wave32 layouts per CDNA5 ISA 7.12.2)
// A (16x32, 16-bit): m = lane%16 ; k = (lane/16)*8 + j  (j<8) / 16+... (j>=8)
// B (32x16, 16-bit): n = lane%16 ; k = (lane/16)*8 + j  (j<8) / 16+... (j>=8)
// C/D (16x16 f32):   n = lane%16 ; m = r + 8*(lane/16)
// ---------------------------------------------------------------------------
__device__ inline v16bf frag_ld_a(const bf16* p0, int ld, int lane) {
  const bf16* p = p0 + (size_t)(lane & 15) * ld + ((lane >> 4) << 3);
  FragBF f;
  f.q[0] = *(const uint4*)(p);
  f.q[1] = *(const uint4*)(p + 16);
  return f.v;
}

__device__ inline v16bf frag_ld_a_clamp(const bf16* base, int ld, int row0,
                                        int maxrow, int col0, int lane) {
  int r = row0 + (lane & 15);
  if (r > maxrow) r = maxrow;
  const bf16* p = base + (size_t)r * ld + col0 + ((lane >> 4) << 3);
  FragBF f;
  f.q[0] = *(const uint4*)(p);
  f.q[1] = *(const uint4*)(p + 16);
  return f.v;
}

__device__ inline v16bf frag_ld_b(const bf16* p0, int ld, int lane) {
  const bf16* p = p0 + (size_t)(lane & 15) * ld + ((lane >> 4) << 3);
  FragBF f;
  f.q[0] = *(const uint4*)(p);
  f.q[1] = *(const uint4*)(p + 16);
  return f.v;
}

__device__ inline v16bf frag_ld_b_clamp(const bf16* base, int ld, int n0,
                                        int maxn, int k0, int lane) {
  int r = n0 + (lane & 15);
  if (r > maxn) r = maxn;
  const bf16* p = base + (size_t)r * ld + k0 + ((lane >> 4) << 3);
  FragBF f;
  f.q[0] = *(const uint4*)(p);
  f.q[1] = *(const uint4*)(p + 16);
  return f.v;
}

__device__ inline v8f wmma_bf16(v16bf a, v16bf b, v8f c) {
  return __builtin_amdgcn_wmma_f32_16x16x32_bf16(
      /*neg_a=*/false, a, /*neg_b=*/false, b,
      /*c_mod=*/(short)0, c, /*reuse_a=*/false, /*reuse_b=*/false);
}

// ---------------------------------------------------------------------------
// Stage 0: fp32 -> bf16 conversions
// ---------------------------------------------------------------------------
__global__ void cvt_f32_bf16(const float* __restrict__ src,
                             bf16* __restrict__ dst, int count) {
  for (int i = blockIdx.x * blockDim.x + threadIdx.x; i < count;
       i += gridDim.x * blockDim.x)
    dst[i] = (bf16)src[i];
}

// dst[(c, r)] = src[(r, c)] : produces B^T row-major in bf16
__global__ void transpose_f32_bf16(const float* __restrict__ src,
                                   bf16* __restrict__ dst, int rows, int cols) {
  int i = blockIdx.x * blockDim.x + threadIdx.x;
  if (i >= rows * cols) return;
  int r = i / cols, c = i % cols;
  dst[(size_t)c * rows + r] = (bf16)src[i];
}

// ---------------------------------------------------------------------------
// GEMM: C = A(MxK) * BT(NxK)^T ; one wave per 32x64 tile, double-buffered.
// M assumed multiple of 32 (32800 = 1025*32). K multiple of 32.
// ---------------------------------------------------------------------------
__global__ __launch_bounds__(32) void gemm_bf16(const bf16* __restrict__ A,
                                                const bf16* __restrict__ BT,
                                                bf16* __restrict__ C,
                                                int N, int K) {
  int lane = threadIdx.x;
  int row0 = blockIdx.x * 32;
  int col0 = blockIdx.y * 64;
  v8f acc[2][4] = {};

  const bf16* A0 = A + (size_t)row0 * K;
  const bf16* A1 = A + (size_t)(row0 + 16) * K;
  const bf16* B0 = BT + (size_t)(col0 + 0)  * K;
  const bf16* B1 = BT + (size_t)(col0 + 16) * K;
  const bf16* B2 = BT + (size_t)(col0 + 32) * K;
  const bf16* B3 = BT + (size_t)(col0 + 48) * K;

  v16bf a0 = frag_ld_a(A0, K, lane);
  v16bf a1 = frag_ld_a(A1, K, lane);
  v16bf b0 = frag_ld_b(B0, K, lane);
  v16bf b1 = frag_ld_b(B1, K, lane);
  v16bf b2 = frag_ld_b(B2, K, lane);
  v16bf b3 = frag_ld_b(B3, K, lane);

  for (int k = 0; k < K; k += 32) {
    int kk = (k + 32 < K) ? (k + 32) : k;   // last iter reloads (discarded)
    v16bf a0n = frag_ld_a(A0 + kk, K, lane);
    v16bf a1n = frag_ld_a(A1 + kk, K, lane);
    v16bf b0n = frag_ld_b(B0 + kk, K, lane);
    v16bf b1n = frag_ld_b(B1 + kk, K, lane);
    v16bf b2n = frag_ld_b(B2 + kk, K, lane);
    v16bf b3n = frag_ld_b(B3 + kk, K, lane);

    acc[0][0] = wmma_bf16(a0, b0, acc[0][0]);
    acc[1][0] = wmma_bf16(a1, b0, acc[1][0]);
    acc[0][1] = wmma_bf16(a0, b1, acc[0][1]);
    acc[1][1] = wmma_bf16(a1, b1, acc[1][1]);
    acc[0][2] = wmma_bf16(a0, b2, acc[0][2]);
    acc[1][2] = wmma_bf16(a1, b2, acc[1][2]);
    acc[0][3] = wmma_bf16(a0, b3, acc[0][3]);
    acc[1][3] = wmma_bf16(a1, b3, acc[1][3]);

    a0 = a0n; a1 = a1n; b0 = b0n; b1 = b1n; b2 = b2n; b3 = b3n;
  }
  int half8 = (lane >> 4) << 3;
  int lcol  = lane & 15;
#pragma unroll
  for (int u = 0; u < 2; ++u)
#pragma unroll
    for (int f = 0; f < 4; ++f)
#pragma unroll
      for (int r = 0; r < 8; ++r)
        C[(size_t)(row0 + u * 16 + r + half8) * N + col0 + f * 16 + lcol] =
            (bf16)acc[u][f][r];
}

// Final projection: fp32 output + bias
__global__ __launch_bounds__(32) void gemm_proj_out(
    const bf16* __restrict__ A, const bf16* __restrict__ BT,
    const float* __restrict__ bias, float* __restrict__ C, int N, int K) {
  int lane = threadIdx.x;
  int row0 = blockIdx.x * 32;
  int col0 = blockIdx.y * 64;
  v8f acc[2][4] = {};

  const bf16* A0 = A + (size_t)row0 * K;
  const bf16* A1 = A + (size_t)(row0 + 16) * K;
  const bf16* B0 = BT + (size_t)(col0 + 0)  * K;
  const bf16* B1 = BT + (size_t)(col0 + 16) * K;
  const bf16* B2 = BT + (size_t)(col0 + 32) * K;
  const bf16* B3 = BT + (size_t)(col0 + 48) * K;

  v16bf a0 = frag_ld_a(A0, K, lane);
  v16bf a1 = frag_ld_a(A1, K, lane);
  v16bf b0 = frag_ld_b(B0, K, lane);
  v16bf b1 = frag_ld_b(B1, K, lane);
  v16bf b2 = frag_ld_b(B2, K, lane);
  v16bf b3 = frag_ld_b(B3, K, lane);

  for (int k = 0; k < K; k += 32) {
    int kk = (k + 32 < K) ? (k + 32) : k;
    v16bf a0n = frag_ld_a(A0 + kk, K, lane);
    v16bf a1n = frag_ld_a(A1 + kk, K, lane);
    v16bf b0n = frag_ld_b(B0 + kk, K, lane);
    v16bf b1n = frag_ld_b(B1 + kk, K, lane);
    v16bf b2n = frag_ld_b(B2 + kk, K, lane);
    v16bf b3n = frag_ld_b(B3 + kk, K, lane);

    acc[0][0] = wmma_bf16(a0, b0, acc[0][0]);
    acc[1][0] = wmma_bf16(a1, b0, acc[1][0]);
    acc[0][1] = wmma_bf16(a0, b1, acc[0][1]);
    acc[1][1] = wmma_bf16(a1, b1, acc[1][1]);
    acc[0][2] = wmma_bf16(a0, b2, acc[0][2]);
    acc[1][2] = wmma_bf16(a1, b2, acc[1][2]);
    acc[0][3] = wmma_bf16(a0, b3, acc[0][3]);
    acc[1][3] = wmma_bf16(a1, b3, acc[1][3]);

    a0 = a0n; a1 = a1n; b0 = b0n; b1 = b1n; b2 = b2n; b3 = b3n;
  }
  int half8 = (lane >> 4) << 3;
  int lcol  = lane & 15;
#pragma unroll
  for (int u = 0; u < 2; ++u)
#pragma unroll
    for (int f = 0; f < 4; ++f)
#pragma unroll
      for (int r = 0; r < 8; ++r) {
        int n = col0 + f * 16 + lcol;
        C[(size_t)(row0 + u * 16 + r + half8) * N + n] =
            acc[u][f][r] + bias[n];
      }
}

// ---------------------------------------------------------------------------
// RMSNorm + head pack.
//   qn, kn : (b, h, n, 64) bf16 ; q has g and SCALE folded in, k has g.
//   vT     : (b, h, 64, NPAD) bf16, zero-padded columns [NSEQ, NPAD)
// ---------------------------------------------------------------------------
__global__ void norm_pack(const bf16* __restrict__ qkv,
                          const float* __restrict__ g,
                          bf16* __restrict__ qn, bf16* __restrict__ kn,
                          bf16* __restrict__ vT) {
  int idx = blockIdx.x * blockDim.x + threadIdx.x;
  int n  = idx % NPAD;
  int bh = idx / NPAD;
  if (bh >= BATCH * HEADS) return;
  if (n >= NSEQ) {  // zero-pad V columns so padded keys contribute nothing
#pragma unroll 8
    for (int d = 0; d < DH; ++d)
      vT[((size_t)bh * DH + d) * NPAD + n] = (bf16)0.f;
    return;
  }
  int h = bh % HEADS, b = bh / HEADS;
  const bf16* base =
      qkv + (size_t)(b * NSEQ + n) * (3 * HEADS * DH) + h * DH;

  union Row { uint4 q4[8]; bf16 e[64]; };
  Row rq, rk, rv, oq, ok;
#pragma unroll
  for (int i = 0; i < 8; ++i) {
    rq.q4[i] = ((const uint4*)(base))[i];
    rk.q4[i] = ((const uint4*)(base + HEADS * DH))[i];
    rv.q4[i] = ((const uint4*)(base + 2 * HEADS * DH))[i];
  }
  float sq = 0.f, sk = 0.f;
#pragma unroll
  for (int d = 0; d < DH; ++d) {
    float tq = (float)rq.e[d];
    float tk = (float)rk.e[d];
    sq += tq * tq;
    sk += tk * tk;
  }
  float rqs = rsqrtf(sq * (1.f / DH) + EPS) * SCALE;
  float rks = rsqrtf(sk * (1.f / DH) + EPS);
#pragma unroll
  for (int d = 0; d < DH; ++d) {
    float gd = g[d];
    oq.e[d] = (bf16)((float)rq.e[d] * rqs * gd);
    ok.e[d] = (bf16)((float)rk.e[d] * rks * gd);
    vT[((size_t)bh * DH + d) * NPAD + n] = rv.e[d];
  }
  uint4* qdst = (uint4*)(qn + ((size_t)bh * NSEQ + n) * DH);
  uint4* kdst = (uint4*)(kn + ((size_t)bh * NSEQ + n) * DH);
#pragma unroll
  for (int i = 0; i < 8; ++i) { qdst[i] = oq.q4[i]; kdst[i] = ok.q4[i]; }
}

// ---------------------------------------------------------------------------
// Flash attention with relative-position bias gather.
// Grid: (33 super-tiles of 32 queries, HEADS, BATCH); one wave each.
// Two 16-query tiles per wave so every K/V fragment feeds 2 WMMAs.
// ---------------------------------------------------------------------------
__global__ __launch_bounds__(32) void flash_attn(
    const bf16* __restrict__ qn, const bf16* __restrict__ kn,
    const bf16* __restrict__ vT, const int* __restrict__ bias_idx,
    const float* __restrict__ pe_table, bf16* __restrict__ attn_out) {
  int lane = threadIdx.x;
  int qt2 = blockIdx.x, h = blockIdx.y, b = blockIdx.z;
  int bh = b * HEADS + h;
  const bf16* Q  = qn + (size_t)bh * NSEQ * DH;
  const bf16* Kp = kn + (size_t)bh * NSEQ * DH;
  const bf16* Vt = vT + (size_t)bh * DH * NPAD;

  __shared__ __align__(16) bf16 ldsP[2][16 * 32];

  int half  = lane >> 4;   // 0 or 1
  int half8 = half << 3;
  int lcol  = lane & 15;
  int qrow0 = qt2 * 32;

  v16bf aq[2][2];
#pragma unroll
  for (int u = 0; u < 2; ++u) {
    aq[u][0] = frag_ld_a_clamp(Q, DH, qrow0 + u * 16, NSEQ - 1, 0, lane);
    aq[u][1] = frag_ld_a_clamp(Q, DH, qrow0 + u * 16, NSEQ - 1, 32, lane);
  }

  float mrun[2][8], lrun[2][8];
  v8f acc[2][4] = {};
#pragma unroll
  for (int u = 0; u < 2; ++u)
#pragma unroll
    for (int r = 0; r < 8; ++r) { mrun[u][r] = -3.0e38f; lrun[u][r] = 0.f; }

  for (int kc = 0; kc < NPAD; kc += 32) {   // 33 key chunks of 32
    // prefetch next chunk of K and V streams (global_prefetch_b8)
    if (kc + 32 < NPAD) {
      __builtin_prefetch(Kp + (size_t)(kc + 32) * DH, 0, 1);
      __builtin_prefetch(Vt + (kc + 32), 0, 1);
    }
    // K fragments shared by both query tiles
    v16bf bk0 = frag_ld_b_clamp(Kp, DH, kc,      NSEQ - 1, 0,  lane);
    v16bf bk1 = frag_ld_b_clamp(Kp, DH, kc,      NSEQ - 1, 32, lane);
    v16bf bk2 = frag_ld_b_clamp(Kp, DH, kc + 16, NSEQ - 1, 0,  lane);
    v16bf bk3 = frag_ld_b_clamp(Kp, DH, kc + 16, NSEQ - 1, 32, lane);

#pragma unroll
    for (int u = 0; u < 2; ++u) {
      // ---- S = q @ k^T : 16x32 scores (two 16x16 D fragments) ----
      v8f s[2];
      {
        v8f s0 = {};
        s0 = wmma_bf16(aq[u][0], bk0, s0);
        s0 = wmma_bf16(aq[u][1], bk1, s0);
        v8f s1 = {};
        s1 = wmma_bf16(aq[u][0], bk2, s1);
        s1 = wmma_bf16(aq[u][1], bk3, s1);
        s[0] = s0; s[1] = s1;
      }
      // ---- relative-position bias gather + key mask ----
#pragma unroll
      for (int t = 0; t < 2; ++t) {
        int j = kc + t * 16 + lcol;          // key index for this lane
        bool jvalid = (j < NSEQ);
#pragma unroll
        for (int r = 0; r < 8; ++r) {
          int i = qrow0 + u * 16 + r + half8;  // query index
          float sv = s[t][r];
          if (jvalid && j >= 1 && i >= 1 && i < NSEQ) {
            int bi = bias_idx[(i - 1) * 1024 + (j - 1)];
            sv += pe_table[bi * HEADS + h];
          }
          if (!jvalid) sv = -3.0e38f;
          s[t][r] = sv;
        }
      }
      // ---- online softmax update (row = fragment element) ----
      float alpha[8];
#pragma unroll
      for (int r = 0; r < 8; ++r) {
        float mx = fmaxf(s[0][r], s[1][r]);
#pragma unroll
        for (int off = 1; off < 16; off <<= 1)
          mx = fmaxf(mx, __shfl_xor(mx, off, 32));
        float mnew = fmaxf(mrun[u][r], mx);
        float a  = __expf(mrun[u][r] - mnew);
        float p0 = __expf(s[0][r] - mnew);
        float p1 = __expf(s[1][r] - mnew);
        float ps = p0 + p1;
#pragma unroll
        for (int off = 1; off < 16; off <<= 1)
          ps += __shfl_xor(ps, off, 32);
        lrun[u][r]  = lrun[u][r] * a + ps;
        mrun[u][r]  = mnew;
        alpha[r] = a;
        s[0][r] = p0;
        s[1][r] = p1;
      }
#pragma unroll
      for (int f = 0; f < 4; ++f)
#pragma unroll
        for (int r = 0; r < 8; ++r)
          acc[u][f][r] *= alpha[r];
      // ---- stash P (D-fragment layout) into LDS ----
#pragma unroll
      for (int t = 0; t < 2; ++t)
#pragma unroll
        for (int r = 0; r < 8; ++r)
          ldsP[u][(r + half8) * 32 + t * 16 + lcol] = (bf16)s[t][r];
    }
    __syncthreads();  // single-wave WG: cheap; orders LDS write -> read
    // ---- reload P as A fragments; O += P @ V with shared V fragments ----
    v16bf ap0 = frag_ld_a(ldsP[0], 32, lane);
    v16bf ap1 = frag_ld_a(ldsP[1], 32, lane);
#pragma unroll
    for (int f = 0; f < 4; ++f) {
      v16bf bv = frag_ld_b(Vt + (size_t)(f * 16) * NPAD + kc, NPAD, lane);
      acc[0][f] = wmma_bf16(ap0, bv, acc[0][f]);
      acc[1][f] = wmma_bf16(ap1, bv, acc[1][f]);
    }
    __syncthreads();
  }
  // ---- epilogue: O / l, store (b, n, h*64 + d) bf16 ----
#pragma unroll
  for (int u = 0; u < 2; ++u)
#pragma unroll
    for (int f = 0; f < 4; ++f)
#pragma unroll
      for (int r = 0; r < 8; ++r) {
        int i = qrow0 + u * 16 + r + half8;
        if (i < NSEQ) {
          float o = acc[u][f][r] / lrun[u][r];
          attn_out[(size_t)(b * NSEQ + i) * (HEADS * DH) + h * DH + f * 16 +
                   lcol] = (bf16)o;
        }
      }
}

// ---------------------------------------------------------------------------
// Host-side orchestration
// ---------------------------------------------------------------------------
extern "C" void kernel_launch(void* const* d_in, const int* in_sizes, int n_in,
                              void* d_out, int out_size, void* d_ws,
                              size_t ws_size, hipStream_t stream) {
  (void)in_sizes; (void)n_in; (void)out_size; (void)ws_size;

  const float* x        = (const float*)d_in[0];
  const float* w_qkv    = (const float*)d_in[1];
  const float* g        = (const float*)d_in[2];
  const float* pe_table = (const float*)d_in[3];
  const float* w_out    = (const float*)d_in[4];
  const float* b_out    = (const float*)d_in[5];
  const int*   bias_idx = (const int*)d_in[6];

  const size_t MROWS = (size_t)BATCH * NSEQ;        // 32800 = 1025 * 32
  const int    MT32  = (int)(MROWS / 32);           // 1025 row tiles

  // workspace layout (bf16 elements, all 16B-aligned)
  bf16* ws      = (bf16*)d_ws;
  bf16* x_bf    = ws;                               // 32800*512
  bf16* wqkvT   = x_bf   + MROWS * DIM;             // 1536*512
  bf16* woutT   = wqkvT  + (size_t)3 * HEADS * DH * DIM;  // 512*512
  bf16* qkv_bf  = woutT  + (size_t)DIM * DIM;       // 32800*1536
  bf16* qn      = qkv_bf + MROWS * 3 * HEADS * DH;  // 256*1025*64
  bf16* kn      = qn     + (size_t)BATCH * HEADS * NSEQ * DH;
  bf16* vT      = kn     + (size_t)BATCH * HEADS * NSEQ * DH; // 256*64*1056
  bf16* attn_o  = qkv_bf;  // alias: qkv_bf dead after norm_pack

  // 1) convert inputs to bf16 (weights pre-transposed -> B^T row-major)
  cvt_f32_bf16<<<4096, 256, 0, stream>>>(x, x_bf, (int)(MROWS * DIM));
  {
    int cnt = DIM * 3 * HEADS * DH;   // 512*1536
    transpose_f32_bf16<<<(cnt + 255) / 256, 256, 0, stream>>>(
        w_qkv, wqkvT, DIM, 3 * HEADS * DH);
  }
  {
    int cnt = DIM * DIM;
    transpose_f32_bf16<<<(cnt + 255) / 256, 256, 0, stream>>>(w_out, woutT,
                                                              DIM, DIM);
  }

  // 2) QKV projection: (32800 x 512) @ (512 x 1536)
  gemm_bf16<<<dim3(MT32, (3 * HEADS * DH) / 64), 32, 0, stream>>>(
      x_bf, wqkvT, qkv_bf, 3 * HEADS * DH, DIM);

  // 3) RMSNorm(q,k)*g (SCALE folded into q) + pack heads, transpose V
  {
    int total = BATCH * HEADS * NPAD;   // 270336
    norm_pack<<<(total + 255) / 256, 256, 0, stream>>>(qkv_bf, g, qn, kn, vT);
  }

  // 4) fused flash attention with relative-position bias
  flash_attn<<<dim3(QT2, HEADS, BATCH), 32, 0, stream>>>(qn, kn, vT, bias_idx,
                                                         pe_table, attn_o);

  // 5) output projection (+ bias), fp32 result
  gemm_proj_out<<<dim3(MT32, DIM / 64), 32, 0, stream>>>(
      attn_o, woutT, b_out, (float*)d_out, DIM, DIM);
}
